// CausalSelfAttention_12369505813015
// MI455X (gfx1250) — matrix operand order repaired and checked by reading, not compiled, not run on hardware
//
#include <hip/hip_runtime.h>

// ---------------------------------------------------------------------------
// CDNA5 (gfx1250) causal self-attention forward.
//  - all matmuls: v_wmma_f32_16x16x32_bf16 (f32 accumulate)
//  - GEMM tiles staged to LDS via TDM (tensor_load_to_lds, 6-arg form),
//    double buffered, one DMA slice per wave
//  - flash-attn K/V tiles staged via global_load_async_to_lds_b128
// ---------------------------------------------------------------------------

typedef __attribute__((ext_vector_type(16))) __bf16 v16bf;
typedef __attribute__((ext_vector_type(8)))  __bf16 v8bf;
typedef __attribute__((ext_vector_type(8)))  float  v8f;
typedef __attribute__((ext_vector_type(4)))  unsigned int v4u;
typedef __attribute__((ext_vector_type(4)))  int v4i;
typedef __attribute__((ext_vector_type(8)))  int v8i;

#define BB 4
#define TT 2048
#define DD 1024
#define HH 16
#define DH 64

#define WMMA_BF16(a, b, c) \
  __builtin_amdgcn_wmma_f32_16x16x32_bf16(false, (a), false, (b), (short)0, (c), false, false)

#if __has_builtin(__builtin_amdgcn_tensor_load_to_lds) && __has_builtin(__builtin_amdgcn_s_wait_tensorcnt)
#define USE_TDM 1
#else
#define USE_TDM 0
#endif

#if __has_builtin(__builtin_amdgcn_global_load_async_to_lds_b128) && __has_builtin(__builtin_amdgcn_s_wait_asynccnt)
#define USE_ASYNC 1
#else
#define USE_ASYNC 0
#endif

static __device__ __forceinline__ v16bf cat8(v8bf lo, v8bf hi) {
  return __builtin_shufflevector(lo, hi, 0, 1, 2, 3, 4, 5, 6, 7, 8, 9, 10, 11, 12, 13, 14, 15);
}

// A-fragment: 16x32 bf16, rows m0..m0+15, cols k0..k0+31 of row-major src (leading dim lda).
// ISA 7.12.2: lane L holds row (L&15); K chunks [kb..kb+7] and [16+kb..16+kb+7], kb=(L>=16)*8.
static __device__ __forceinline__ v16bf frag_a(const __bf16* A, int lda, int m0, int k0, int lane) {
  const __bf16* p = A + (size_t)(m0 + (lane & 15)) * lda + k0;
  int kb = (lane >> 4) << 3;
  v8bf lo = *(const v8bf*)(p + kb);
  v8bf hi = *(const v8bf*)(p + 16 + kb);
  return cat8(lo, hi);
}

// B-fragment: 32x16, element (k,n) = W[n0+n][k0+k], W row-major (leading dim ldw).
// Lane L holds column (L&15); K chunks [kb..kb+7] and [16+kb..16+kb+7], kb=(L>=16)*8.
static __device__ __forceinline__ v16bf frag_b(const __bf16* W, int ldw, int k0, int n0, int lane) {
  const __bf16* p = W + (size_t)(n0 + (lane & 15)) * ldw + k0 + ((lane >> 4) << 3);
  v8bf lo = *(const v8bf*)(p);
  v8bf hi = *(const v8bf*)(p + 16);
  return cat8(lo, hi);
}

// A-fragment from LDS tile with row stride 48 elements (96B -> 16B-aligned chunks).
static __device__ __forceinline__ v16bf frag_a_lds(const __bf16* P, int lane) {
  const __bf16* p = P + (lane & 15) * 48;
  int kb = (lane >> 4) << 3;
  v8bf lo = *(const v8bf*)(p + kb);
  v8bf hi = *(const v8bf*)(p + 16 + kb);
  return cat8(lo, hi);
}

// ---------------------------------------------------------------------------
// TDM: DMA a rows x cols bf16 tile (row stride = stride elements) into LDS.
// D# built per CDNA5 ISA §8.3/§8.4.  All descriptor inputs must be wave-uniform.
// ---------------------------------------------------------------------------
static __device__ __forceinline__ void stage_tile_tdm(__bf16* lptr, const __bf16* gptr,
                                                      int rows, int cols, int stride, int lane) {
#if USE_TDM
  unsigned long long ga = (unsigned long long)(size_t)(const void*)gptr;
  unsigned lds = (unsigned)(size_t)(void*)lptr;
  v4u g0;
  g0.x = 1u;                                                 // count=1, user D#, no gather
  g0.y = lds;                                                // LDS byte address
  g0.z = (unsigned)ga;                                       // global_addr[31:0]
  g0.w = (unsigned)((ga >> 32) & 0x01FFFFFFu) | (2u << 30);  // global_addr[56:32], type=2
  v8i g1;
  g1[0] = 1 << 16;                                           // data_size=1 (2 bytes)
  g1[1] = (cols & 0xFFFF) << 16;                             // tensor_dim0[15:0]
  g1[2] = ((cols >> 16) & 0xFFFF) | ((rows & 0xFFFF) << 16); // dim0[31:16], dim1[15:0]
  g1[3] = ((rows >> 16) & 0xFFFF) | ((cols & 0xFFFF) << 16); // dim1[31:16], tile_dim0
  g1[4] = rows & 0xFFFF;                                     // tile_dim1 (tile_dim2=0)
  g1[5] = stride;                                            // tensor_dim0_stride[31:0]
  g1[6] = 0;                                                 // stride[47:32], dim1_stride lo
  g1[7] = 0;
  v4i g2 = {0, 0, 0, 0};
  v4i g3 = {0, 0, 0, 0};
  v8i g4 = {0, 0, 0, 0, 0, 0, 0, 0};
  __builtin_amdgcn_tensor_load_to_lds(g0, g1, g2, g3, g4, 0);
  (void)lane;
#else
  // wave-cooperative fallback, 16B chunks
  int chunks = (rows * cols) >> 3;
  for (int c = lane; c < chunks; c += 32) {
    int r = (c << 3) / cols;
    int o = (c << 3) % cols;
    *(v8bf*)(lptr + r * cols + o) = *(const v8bf*)(gptr + (size_t)r * stride + o);
  }
#endif
}

static __device__ __forceinline__ void wait_tiles() {
#if USE_TDM
  __builtin_amdgcn_s_wait_tensorcnt(0);
#endif
}

// ---------------------------------------------------------------------------
// Async copy (per-wave): rows x cols bf16, row stride ldg elements, into LDS.
// ---------------------------------------------------------------------------
static __device__ __forceinline__ void stage_async(__bf16* lptr, const __bf16* gptr,
                                                   int rows, int cols, int ldg, int lane) {
  int cpr = cols >> 3;  // 16B chunks per row
  int total = rows * cpr;
#if USE_ASYNC
  typedef __attribute__((address_space(1))) v4i GAS;
  typedef __attribute__((address_space(3))) v4i LAS;
  for (int c = lane; c < total; c += 32) {
    int r = c / cpr;
    int o = (c % cpr) << 3;
    GAS* g = (GAS*)(size_t)(const void*)(gptr + (size_t)r * ldg + o);
    LAS* l = (LAS*)(unsigned)(size_t)(void*)(lptr + r * cols + o);
    __builtin_amdgcn_global_load_async_to_lds_b128(g, l, 0, 0);
  }
#else
  for (int c = lane; c < total; c += 32) {
    int r = c / cpr;
    int o = (c % cpr) << 3;
    *(v8bf*)(lptr + r * cols + o) = *(const v8bf*)(gptr + (size_t)r * ldg + o);
  }
#endif
}

static __device__ __forceinline__ void wait_async_done() {
#if USE_ASYNC
  __builtin_amdgcn_s_wait_asynccnt(0);
#endif
}

// ---------------------------------------------------------------------------
// fp32 -> bf16 convert
// ---------------------------------------------------------------------------
__global__ void cvt_f32_bf16(const float* __restrict__ s, __bf16* __restrict__ d, size_t n) {
  size_t i = (size_t)blockIdx.x * blockDim.x + threadIdx.x;
  if (i < n) d[i] = (__bf16)s[i];
}

// ---------------------------------------------------------------------------
// V transpose: Vb [B*T, D] (col = h*DH+d)  ->  Vt [B, H, DH, T]
// ---------------------------------------------------------------------------
__global__ void transpose_v(const __bf16* __restrict__ Vb, __bf16* __restrict__ Vt) {
  size_t i = (size_t)blockIdx.x * blockDim.x + threadIdx.x;
  if (i >= (size_t)BB * HH * DH * TT) return;
  int t = (int)(i % TT);
  size_t r = i / TT;
  int d = (int)(r % DH);
  r /= DH;
  int h = (int)(r % HH);
  int b = (int)(r / HH);
  Vt[i] = Vb[((size_t)(b * TT + t)) * DD + h * DH + d];
}

// ---------------------------------------------------------------------------
// GEMM: C[M,N] = A[M,K] x W[N,K]^T.  Block 256 thr = 8 waves; block tile
// 128x128; wave tile 32x64; K-step 64.  A/W tiles TDM-staged, double buffered:
// waves 0-3 DMA 32-row slices of the A tile, waves 4-7 the W tile.
// ---------------------------------------------------------------------------
template <typename OutT>
__global__ __launch_bounds__(256) void gemm_xwT(const __bf16* __restrict__ A,
                                                const __bf16* __restrict__ W,
                                                OutT* __restrict__ C, int M, int N, int K) {
  __shared__ __attribute__((aligned(16))) __bf16 Abuf[2][128 * 64];
  __shared__ __attribute__((aligned(16))) __bf16 Wbuf[2][128 * 64];

  int lane = threadIdx.x & 31;
  int wave = __builtin_amdgcn_readfirstlane(threadIdx.x >> 5);  // scalar for TDM descriptors
  int mB = blockIdx.x * 128;
  int nB = blockIdx.y * 128;
  int mw = (wave >> 1) * 32;
  int nw = (wave & 1) * 64;

  int r0 = (wave & 3) * 32;
  const __bf16* gA = A + (size_t)(mB + r0) * K;
  const __bf16* gW = W + (size_t)(nB + r0) * K;

  // prologue: fill buffer 0
  if (wave < 4) stage_tile_tdm(&Abuf[0][r0 * 64], gA, 32, 64, K, lane);
  else          stage_tile_tdm(&Wbuf[0][r0 * 64], gW, 32, 64, K, lane);
  wait_tiles();
  __syncthreads();

  v8f acc[2][4] = {};
  int nkt = K / 64;
  for (int kt = 0; kt < nkt; ++kt) {
    int cur = kt & 1;
    if (kt + 1 < nkt) {
      int nxt = cur ^ 1;
      int kg = (kt + 1) * 64;
      if (wave < 4) stage_tile_tdm(&Abuf[nxt][r0 * 64], gA + kg, 32, 64, K, lane);
      else          stage_tile_tdm(&Wbuf[nxt][r0 * 64], gW + kg, 32, 64, K, lane);
    }
#pragma unroll
    for (int kk = 0; kk < 64; kk += 32) {
      v16bf a0 = frag_a(&Abuf[cur][0], 64, mw, kk, lane);
      v16bf a1 = frag_a(&Abuf[cur][0], 64, mw + 16, kk, lane);
#pragma unroll
      for (int j = 0; j < 4; ++j) {
        v16bf b = frag_b(&Wbuf[cur][0], 64, kk, nw + 16 * j, lane);
        acc[0][j] = WMMA_BF16(a0, b, acc[0][j]);
        acc[1][j] = WMMA_BF16(a1, b, acc[1][j]);
      }
    }
    wait_tiles();      // next buffer's DMA done (this wave's TENSORcnt)
    __syncthreads();   // everyone done computing + DMA landed
  }

  int cn = lane & 15;
  int hl = lane >> 4;
#pragma unroll
  for (int i = 0; i < 2; ++i)
#pragma unroll
    for (int j = 0; j < 4; ++j) {
      int n = nB + nw + 16 * j + cn;
#pragma unroll
      for (int v = 0; v < 8; ++v) {
        int m = mB + mw + 16 * i + v + 8 * hl;
        C[(size_t)m * N + n] = (OutT)acc[i][j][v];
      }
    }
}

// ---------------------------------------------------------------------------
// Flash attention (causal, online softmax).
// Grid (T/64, H, B); block 128 thr = 4 waves; each wave owns 16 query rows and
// its own causal KV range in steps of 32 keys.  K/V tiles async-staged to
// per-wave LDS; P staged through LDS for the C-layout -> A-layout relayout.
// ---------------------------------------------------------------------------
__global__ __launch_bounds__(128) void flash_attn(const __bf16* __restrict__ Qg,
                                                  const __bf16* __restrict__ Kg,
                                                  const __bf16* __restrict__ Vt,
                                                  __bf16* __restrict__ Ctx) {
  __shared__ __attribute__((aligned(16))) __bf16 KtileS[4][32 * 64];
  __shared__ __attribute__((aligned(16))) __bf16 VtileS[4][64 * 32];
  __shared__ __attribute__((aligned(16))) __bf16 Plds[4][16][48];

  int lane = threadIdx.x & 31;
  int wave = threadIdx.x >> 5;
  int b = blockIdx.z;
  int h = blockIdx.y;
  int q0 = blockIdx.x * 64 + wave * 16;

  const __bf16* Qh = Qg + (size_t)b * TT * DD + h * DH;
  const __bf16* Kh = Kg + (size_t)b * TT * DD + h * DH;
  const __bf16* Vh = Vt + (size_t)(b * HH + h) * DH * TT;
  __bf16* Kt = &KtileS[wave][0];
  __bf16* Vw = &VtileS[wave][0];
  __bf16* Pw = &Plds[wave][0][0];

  v16bf qa0 = frag_a(Qh, DD, q0, 0, lane);
  v16bf qa1 = frag_a(Qh, DD, q0, 32, lane);

  v8f O[4] = {};
  float m_i[8], l_i[8];
#pragma unroll
  for (int v = 0; v < 8; ++v) {
    m_i[v] = -1.0e30f;
    l_i[v] = 0.0f;
  }

  int cn = lane & 15;
  int hl = lane >> 4;
  const float scale = 0.125f;  // 1/sqrt(64)

  for (int s = 0; s <= q0 + 15; s += 32) {
#if USE_ASYNC
    // close the WAR window: previous step's ds reads of Kt/Vw must retire
    // before the async DMA overwrites them (async is unordered vs DS).
    asm volatile("s_wait_dscnt 0" ::: "memory");
#endif
    stage_async(Kt, Kh + (size_t)s * DD, 32, 64, DD, lane);  // 32 keys x dh=64
    stage_async(Vw, Vh + s, 64, 32, TT, lane);               // dh=64 rows x 32 keys
    wait_async_done();

    // ---- S = (Q K^T) * scale : two 16x16 f32 frags over 32 keys ----
    v8f S[2] = {};
#pragma unroll
    for (int nt = 0; nt < 2; ++nt) {
      v16bf bk0 = frag_b(Kt, 64, 0, 16 * nt, lane);
      v16bf bk1 = frag_b(Kt, 64, 32, 16 * nt, lane);
      S[nt] = WMMA_BF16(qa0, bk0, S[nt]);
      S[nt] = WMMA_BF16(qa1, bk1, S[nt]);
    }

    // ---- causal mask + online softmax ----
#pragma unroll
    for (int v = 0; v < 8; ++v) {
      int row = q0 + v + 8 * hl;
      float x0 = S[0][v] * scale;
      float x1 = S[1][v] * scale;
      if (s + cn > row) x0 = -1.0e30f;
      if (s + 16 + cn > row) x1 = -1.0e30f;

      float mx = fmaxf(x0, x1);
#pragma unroll
      for (int off = 1; off < 16; off <<= 1) mx = fmaxf(mx, __shfl_xor(mx, off, 32));
      float mnew = fmaxf(m_i[v], mx);

      float a = __expf(m_i[v] - mnew);
      m_i[v] = mnew;
      float p0 = __expf(x0 - mnew);
      float p1 = __expf(x1 - mnew);

      Plds[wave][v + 8 * hl][cn] = (__bf16)p0;
      Plds[wave][v + 8 * hl][16 + cn] = (__bf16)p1;

      float rs = p0 + p1;
#pragma unroll
      for (int off = 1; off < 16; off <<= 1) rs += __shfl_xor(rs, off, 32);
      l_i[v] = l_i[v] * a + rs;
#pragma unroll
      for (int j = 0; j < 4; ++j) O[j][v] *= a;
    }

    // ---- O += P x V ----
    v16bf pa = frag_a_lds(Pw, lane);
#pragma unroll
    for (int j = 0; j < 4; ++j) {
      v16bf bv = frag_b(Vw, 32, 0, 16 * j, lane);
      O[j] = WMMA_BF16(pa, bv, O[j]);
    }
  }

  // ---- epilogue: normalize, store context bf16 [B*T, D] ----
  float inv[8];
#pragma unroll
  for (int v = 0; v < 8; ++v) inv[v] = 1.0f / l_i[v];
#pragma unroll
  for (int j = 0; j < 4; ++j) {
    int d = 16 * j + cn;
#pragma unroll
    for (int v = 0; v < 8; ++v) {
      int q = q0 + v + 8 * hl;
      Ctx[((size_t)(b * TT + q)) * DD + h * DH + d] = (__bf16)(O[j][v] * inv[v]);
    }
  }
}

// ---------------------------------------------------------------------------
// Host-side orchestration
// ---------------------------------------------------------------------------
extern "C" void kernel_launch(void* const* d_in, const int* in_sizes, int n_in,
                              void* d_out, int out_size, void* d_ws, size_t ws_size,
                              hipStream_t stream) {
  const float* x  = (const float*)d_in[0];
  const float* Wq = (const float*)d_in[1];
  const float* Wk = (const float*)d_in[2];
  const float* Wv = (const float*)d_in[3];
  const float* Wo = (const float*)d_in[4];

  const size_t SZW = (size_t)DD * DD;
  const size_t MT  = (size_t)BB * TT;
  const size_t SZX = MT * DD;

  __bf16* ws  = (__bf16*)d_ws;
  __bf16* Wqb = ws;
  __bf16* Wkb = Wqb + SZW;
  __bf16* Wvb = Wkb + SZW;
  __bf16* Wob = Wvb + SZW;
  __bf16* Xb  = Wob + SZW;
  __bf16* Qb  = Xb + SZX;
  __bf16* Kb  = Qb + SZX;
  __bf16* Vb  = Kb + SZX;
  __bf16* Vt  = Vb + SZX;
  __bf16* Ctx = Xb;  // X dead after QKV GEMMs; reuse.

  const int thr = 256;
  cvt_f32_bf16<<<(unsigned)((SZX + thr - 1) / thr), thr, 0, stream>>>(x, Xb, SZX);
  cvt_f32_bf16<<<(unsigned)((SZW + thr - 1) / thr), thr, 0, stream>>>(Wq, Wqb, SZW);
  cvt_f32_bf16<<<(unsigned)((SZW + thr - 1) / thr), thr, 0, stream>>>(Wk, Wkb, SZW);
  cvt_f32_bf16<<<(unsigned)((SZW + thr - 1) / thr), thr, 0, stream>>>(Wv, Wvb, SZW);
  cvt_f32_bf16<<<(unsigned)((SZW + thr - 1) / thr), thr, 0, stream>>>(Wo, Wob, SZW);

  dim3 ggrid((unsigned)(MT / 128), (unsigned)(DD / 128));
  gemm_xwT<__bf16><<<ggrid, 256, 0, stream>>>(Xb, Wqb, Qb, (int)MT, DD, DD);
  gemm_xwT<__bf16><<<ggrid, 256, 0, stream>>>(Xb, Wkb, Kb, (int)MT, DD, DD);
  gemm_xwT<__bf16><<<ggrid, 256, 0, stream>>>(Xb, Wvb, Vb, (int)MT, DD, DD);

  transpose_v<<<(unsigned)((SZX + thr - 1) / thr), thr, 0, stream>>>(Vb, Vt);

  dim3 fgrid(TT / 64, HH, BB);
  flash_attn<<<fgrid, 128, 0, stream>>>(Qb, Kb, Vt, Ctx);

  gemm_xwT<float><<<ggrid, 256, 0, stream>>>(Ctx, Wob, (float*)d_out, (int)MT, DD, DD);
}
